// dLSTM_20581483283024
// MI455X (gfx1250) — matrix operand order repaired and checked by reading, not compiled, not run on hardware
//
#include <hip/hip_runtime.h>

// ---------------------------------------------------------------------------
// dLSTM for MI455X (gfx1250): 2-layer LSTM, B=32, H=2048, seq_len from out_size.
// bf16 weights stay resident in the 192MB L2 across all 128 steps; gates GEMM
// via v_wmma_f32_16x16x32_bf16 with K=4096=[x|h]; A chunks staged into LDS
// with double-buffered global_load_async_to_lds_b128 (ASYNCcnt pipeline).
// ---------------------------------------------------------------------------

typedef __attribute__((ext_vector_type(16))) __bf16 v16bf;
typedef __attribute__((ext_vector_type(8)))  float  v8f;

union FragBF {
    v16bf v;
    uint4 q[2];
};

__device__ __forceinline__ unsigned short f2bf(float x) {
    unsigned int u = __float_as_uint(x);
    u += 0x7FFFu + ((u >> 16) & 1u);          // round-to-nearest-even
    return (unsigned short)(u >> 16);
}

__device__ __forceinline__ float sigm(float x) {
    return 1.0f / (1.0f + __expf(-x));
}

__device__ __forceinline__ float tanh_fast(float x) {
    x = fminf(fmaxf(x, -12.0f), 12.0f);
    float e = __expf(2.0f * x);
    return (e - 1.0f) / (e + 1.0f);
}

// Async 16-byte global->LDS copy (per-lane), tracked with ASYNCcnt.
__device__ __forceinline__ void async_stage16(const unsigned short* src,
                                              unsigned short* lds_dst) {
    const unsigned lds_off = (unsigned)(size_t)lds_dst;   // aperture: low 32 bits = LDS offset
    asm volatile("global_load_async_to_lds_b128 %0, %1, off"
                 :: "v"(lds_off), "v"(src)
                 : "memory");
}

__device__ __forceinline__ void wait_async() {
    asm volatile("s_wait_asynccnt 0" ::: "memory");
}

// B fragment (32x16 bf16, K x N): lane L -> column N = L%16, K = (L/16)*16 .. +15
__device__ __forceinline__ v16bf load_b_frag(const unsigned short* __restrict__ wrow, int kb) {
    FragBF f;
    const uint4* p = reinterpret_cast<const uint4*>(wrow + kb);
    f.q[0] = p[0];
    f.q[1] = p[1];
    return f.v;
}

// A fragment (16x32 bf16, M x K) from an LDS chunk laid out [32][128]:
// lane L -> row M, base = (L<16 ? 0 : 16); K = kk+base .. kk+base+15 contiguous.
__device__ __forceinline__ v16bf load_a_frag(const unsigned short* __restrict__ alds,
                                             int m, int kk, int lane) {
    int base = (lane < 16) ? 0 : 16;
    const uint4* p = reinterpret_cast<const uint4*>(alds + m * 128 + kk + base);
    FragBF f;
    f.q[0] = p[0];
    f.q[1] = p[1];   // +16 bytes = +8 bf16 elements
    return f.v;
}

// ---------------------------------------------------------------------------
// Fused LSTM layer step: gates = [A_lo | A_hi] @ W^T + bias ; then cell update.
// Grid: 64 blocks (h-tile = 32), block: 256 threads (8 waves).
// Wave w: gate = w/2, sub = w%2 -> 16 output columns; 2 M-tiles (B=32).
// ---------------------------------------------------------------------------
__global__ __launch_bounds__(256)
void lstm_layer_kernel(const unsigned short* __restrict__ A_lo,   // [32][2048] bf16 (layer input)
                       const unsigned short* __restrict__ A_hi,   // [32][2048] bf16 (h_prev)
                       const unsigned short* __restrict__ W,      // [8192][4096] bf16 ([W_ih|W_hh])
                       const float*          __restrict__ bias,   // [8192] (b_ih+b_hh)
                       float*                __restrict__ c,      // [32][2048] fp32 in/out
                       unsigned short*       __restrict__ h_out_bf,  // [32][2048] bf16
                       float*                __restrict__ h_out_f32) // [32][2048] fp32 or null
{
    __shared__ __align__(16) unsigned short Alds[2][32 * 128];   // 2 x 8 KB (double buffer)
    __shared__ __align__(16) float          Glds[32 * 132];      // gate staging (padded)

    const int tid  = threadIdx.x;
    const int wave = tid >> 5;
    const int lane = tid & 31;
    const int h0   = blockIdx.x * 32;
    const int gate = wave >> 1;
    const int sub  = wave & 1;
    const int ncol = gate * 2048 + h0 + sub * 16 + (lane & 15);  // output gate-row index
    const unsigned short* wrow = W + (size_t)ncol * 4096;

    v8f acc0 = {};
    v8f acc1 = {};

    const int row = tid >> 3;     // A staging: 32 rows x 8 segs of 16 bf16
    const int seg = tid & 7;
    const int lds_slot = row * 128 + seg * 16;

    // prologue: async-stage chunk 0 into buffer 0
    async_stage16(A_lo + row * 2048 + seg * 16, &Alds[0][lds_slot]);
    wait_async();
    __syncthreads();

    int cur = 0;
    for (int kc = 0; kc < 4096; kc += 128) {
        // kick off async staging of the next chunk into the other buffer
        if (kc + 128 < 4096) {
            const int kn = kc + 128;
            const unsigned short* src = (kn < 2048) ? A_lo : A_hi;
            async_stage16(src + row * 2048 + (kn & 2047) + seg * 16,
                          &Alds[cur ^ 1][lds_slot]);
            __builtin_prefetch(wrow + kn + ((lane >> 4) << 4), 0, 1);
        }

        const unsigned short* a = Alds[cur];
#pragma unroll
        for (int kk = 0; kk < 128; kk += 32) {
            const int kb = kc + kk + ((lane >> 4) << 4);
            v16bf bf = load_b_frag(wrow, kb);
            v16bf a0 = load_a_frag(a, (lane & 15),      kk, lane);
            v16bf a1 = load_a_frag(a, 16 + (lane & 15), kk, lane);
            acc0 = __builtin_amdgcn_wmma_f32_16x16x32_bf16(false, a0, false, bf,
                                                           (short)0, acc0, false, false);
            acc1 = __builtin_amdgcn_wmma_f32_16x16x32_bf16(false, a1, false, bf,
                                                           (short)0, acc1, false, false);
        }

        wait_async();        // next chunk landed in LDS
        __syncthreads();     // all waves done reading buf[cur]
        cur ^= 1;
    }

    // ---- bias + spill gate tiles to LDS (C layout: VGPR v -> M = v + 8*(lane/16)) ----
    const float bv  = bias[ncol];
    const int   col = wave * 16 + (lane & 15);     // = gate*32 + h_local
    const int   rb  = (lane >> 4) * 8;
#pragma unroll
    for (int v = 0; v < 8; v++) {
        Glds[(rb + v) * 132 + col]      = acc0[v] + bv;
        Glds[(16 + rb + v) * 132 + col] = acc1[v] + bv;
    }
    __syncthreads();

    // ---- fused LSTM cell elementwise: 32 batch x 32 h per block ----
    for (int e = tid; e < 32 * 32; e += 256) {
        const int b  = e >> 5;
        const int hl = e & 31;
        const float iv = Glds[b * 132 + hl];
        const float fv = Glds[b * 132 + 32 + hl];
        const float gv = Glds[b * 132 + 64 + hl];
        const float ov = Glds[b * 132 + 96 + hl];
        const int gi = b * 2048 + h0 + hl;
        const float cn = sigm(fv) * c[gi] + sigm(iv) * tanh_fast(gv);
        const float hn = sigm(ov) * tanh_fast(cn);
        c[gi] = cn;
        h_out_bf[gi] = f2bf(hn);
        if (h_out_f32) h_out_f32[gi] = hn;
    }
}

// ---------------------------------------------------------------------------
// Projection: x_next = h_top @ W_out^T + b_out  -> bf16 feedback input.
// Grid: 16 blocks x 256 threads; wave w -> 16 columns; K = 2048.
// ---------------------------------------------------------------------------
__global__ __launch_bounds__(256)
void proj_kernel(const unsigned short* __restrict__ A,     // [32][2048] bf16
                 const unsigned short* __restrict__ W,     // [2048][2048] bf16
                 const float*          __restrict__ bias,  // [2048]
                 unsigned short*       __restrict__ out_bf)// [32][2048] bf16
{
    __shared__ __align__(16) unsigned short Alds[2][32 * 128];

    const int tid  = threadIdx.x;
    const int wave = tid >> 5;
    const int lane = tid & 31;
    const int ncol = blockIdx.x * 128 + wave * 16 + (lane & 15);
    const unsigned short* wrow = W + (size_t)ncol * 2048;

    v8f acc0 = {};
    v8f acc1 = {};

    const int row = tid >> 3;
    const int seg = tid & 7;
    const int lds_slot = row * 128 + seg * 16;

    async_stage16(A + row * 2048 + seg * 16, &Alds[0][lds_slot]);
    wait_async();
    __syncthreads();

    int cur = 0;
    for (int kc = 0; kc < 2048; kc += 128) {
        if (kc + 128 < 2048) {
            async_stage16(A + row * 2048 + (kc + 128) + seg * 16,
                          &Alds[cur ^ 1][lds_slot]);
            __builtin_prefetch(wrow + kc + 128 + ((lane >> 4) << 4), 0, 1);
        }

        const unsigned short* a = Alds[cur];
#pragma unroll
        for (int kk = 0; kk < 128; kk += 32) {
            const int kb = kc + kk + ((lane >> 4) << 4);
            v16bf bf = load_b_frag(wrow, kb);
            v16bf a0 = load_a_frag(a, (lane & 15),      kk, lane);
            v16bf a1 = load_a_frag(a, 16 + (lane & 15), kk, lane);
            acc0 = __builtin_amdgcn_wmma_f32_16x16x32_bf16(false, a0, false, bf,
                                                           (short)0, acc0, false, false);
            acc1 = __builtin_amdgcn_wmma_f32_16x16x32_bf16(false, a1, false, bf,
                                                           (short)0, acc1, false, false);
        }

        wait_async();
        __syncthreads();
        cur ^= 1;
    }

    const float bv = bias[ncol];
    const int   rb = (lane >> 4) * 8;
#pragma unroll
    for (int v = 0; v < 8; v++) {
        out_bf[(size_t)(rb + v) * 2048 + ncol]      = f2bf(acc0[v] + bv);
        out_bf[(size_t)(16 + rb + v) * 2048 + ncol] = f2bf(acc1[v] + bv);
    }
}

// ---------------------------------------------------------------------------
// Prep: fp32 -> bf16 weight conversion ([W_ih|W_hh] concatenated along K),
// W_out conversion, combined bias, state init.
// ---------------------------------------------------------------------------
__global__ void build_weights_kernel(const float* __restrict__ W_ih,  // [2][8192][2048]
                                     const float* __restrict__ W_hh,  // [2][8192][2048]
                                     unsigned short* __restrict__ Wbf)// [2][8192][4096]
{
    const long long total = 2LL * 8192 * 4096;   // 2^26
    for (long long idx = (long long)blockIdx.x * 256 + threadIdx.x; idx < total;
         idx += (long long)gridDim.x * 256) {
        const int l   = (int)(idx >> 25);              // 8192*4096 = 2^25
        const int rem = (int)(idx & ((1 << 25) - 1));
        const int n   = rem >> 12;                     // /4096
        const int k   = rem & 4095;
        const size_t base = ((size_t)l << 24) + ((size_t)n << 11);  // l*8192*2048 + n*2048
        const float v = (k < 2048) ? W_ih[base + k] : W_hh[base + (k - 2048)];
        Wbf[idx] = f2bf(v);
    }
}

__global__ void build_wout_kernel(const float* __restrict__ W_out,    // [2048][2048]
                                  unsigned short* __restrict__ Wbf)   // [2048][2048]
{
    const int total = 2048 * 2048;
    for (int idx = blockIdx.x * 256 + threadIdx.x; idx < total; idx += gridDim.x * 256)
        Wbf[idx] = f2bf(W_out[idx]);
}

__global__ void init_state_kernel(const float* __restrict__ h0,    // [2][32][2048]
                                  const float* __restrict__ c0,    // [2][32][2048]
                                  const float* __restrict__ b_ih,  // [2][8192]
                                  const float* __restrict__ b_hh,  // [2][8192]
                                  unsigned short* __restrict__ h_bf,  // [2][2][32*2048]
                                  float* __restrict__ c_ws,           // [2][32*2048]
                                  float* __restrict__ biasc,          // [2][8192]
                                  unsigned short* __restrict__ x_bf)  // [32*2048]
{
    const int stride = gridDim.x * 256;
    const int t0 = blockIdx.x * 256 + threadIdx.x;
    for (int idx = t0; idx < 2 * 65536; idx += stride) {     // states
        const int l  = idx >> 16;
        const int rc = idx & 65535;
        h_bf[(l * 2 + 0) * 65536 + rc] = f2bf(h0[idx]);
        c_ws[idx] = c0[idx];
    }
    for (int idx = t0; idx < 2 * 8192; idx += stride)        // combined bias
        biasc[idx] = b_ih[idx] + b_hh[idx];
    for (int idx = t0; idx < 65536; idx += stride)           // x0 = 0
        x_bf[idx] = 0;
}

// ---------------------------------------------------------------------------
extern "C" void kernel_launch(void* const* d_in, const int* in_sizes, int n_in,
                              void* d_out, int out_size, void* d_ws, size_t ws_size,
                              hipStream_t stream) {
    (void)in_sizes; (void)n_in; (void)ws_size;

    const float* h0    = (const float*)d_in[0];
    const float* c0    = (const float*)d_in[1];
    const float* W_ih  = (const float*)d_in[2];
    const float* W_hh  = (const float*)d_in[3];
    const float* b_ih  = (const float*)d_in[4];
    const float* b_hh  = (const float*)d_in[5];
    const float* W_out = (const float*)d_in[6];
    const float* b_out = (const float*)d_in[7];

    // workspace layout (all offsets 256B-aligned by construction)
    char* ws = (char*)d_ws;
    unsigned short* Wbf    = (unsigned short*)(ws);                      // 2*8192*4096*2  = 128 MB
    unsigned short* Woutbf = (unsigned short*)(ws + 134217728);          // 2048*2048*2    = 8 MB
    float*          biasc  = (float*)(ws + 134217728 + 8388608);         // 2*8192*4       = 64 KB
    unsigned short* h_bf   = (unsigned short*)(ws + 134217728 + 8388608 + 65536);        // 512 KB
    float*          c_ws   = (float*)(ws + 134217728 + 8388608 + 65536 + 524288);        // 512 KB
    unsigned short* x_bf   = (unsigned short*)(ws + 134217728 + 8388608 + 65536 + 1048576); // 128 KB

    // one-time (per launch) conversions — deterministic, inputs never mutated
    build_weights_kernel<<<2048, 256, 0, stream>>>(W_ih, W_hh, Wbf);
    build_wout_kernel<<<512, 256, 0, stream>>>(W_out, Woutbf);
    init_state_kernel<<<512, 256, 0, stream>>>(h0, c0, b_ih, b_hh, h_bf, c_ws, biasc, x_bf);

    const int BH = 32 * 2048;                 // 65536
    const int steps = out_size / BH;          // = seq_len (128)
    float* out = (float*)d_out;

    int cur = 0;
    for (int t = 0; t < steps; ++t) {
        const int nxt = cur ^ 1;
        // layer 0: A = [x | h0_prev]
        lstm_layer_kernel<<<64, 256, 0, stream>>>(
            x_bf, h_bf + (0 * 2 + cur) * BH,
            Wbf, biasc, c_ws,
            h_bf + (0 * 2 + nxt) * BH, nullptr);
        // layer 1: A = [h0_new | h1_prev]; top-layer h -> d_out[t]
        lstm_layer_kernel<<<64, 256, 0, stream>>>(
            h_bf + (0 * 2 + nxt) * BH, h_bf + (1 * 2 + cur) * BH,
            Wbf + (size_t)8192 * 4096, biasc + 8192, c_ws + BH,
            h_bf + (1 * 2 + nxt) * BH, out + (size_t)t * BH);
        // feedback projection: x_{t+1} = h_top @ W_out^T + b_out (bf16)
        proj_kernel<<<16, 256, 0, stream>>>(
            h_bf + (1 * 2 + nxt) * BH, Woutbf, b_out, x_bf);
        cur = nxt;
    }
}
